// Model_E_UPMiM_1649267442034
// MI455X (gfx1250) — matrix operand order repaired and checked by reading, not compiled, not run on hardware
//
#include <hip/hip_runtime.h>
#include <hip/hip_bf16.h>

// ---------------------------------------------------------------------------
// Model_E_UPMiM (MIND capsule interest) for gfx1250.
//  - adj / hat GEMMs on v_wmma_f32_16x16x32_bf16 (bf16 staged operands,
//    f32 accumulate): workload is HBM-bound (~0.5GB traffic @ 23.3TB/s),
//    so halving staged bytes + fast WMMA path is the right trade.
//  - routing holds all of hat[b] (100KB bf16) in the 320KB WGP LDS, loaded
//    via global_load_async_to_lds_b128 (ASYNCcnt-tracked async path).
// ---------------------------------------------------------------------------

typedef __attribute__((ext_vector_type(16))) __bf16 v16bf;
typedef __attribute__((ext_vector_type(8)))  __bf16 v8bf;
typedef __attribute__((ext_vector_type(8)))  float  v8f;

#define B_   1024
#define S_   200
#define SPAD 208
#define D_   64
#define NI_  4
#define E_   256   // NI*D

// ---- WMMA fragment loaders (CDNA5 ISA 7.12.2 16-bit layouts) ---------------
// A 16x32 (MxK): lane = {half,m}; element e <-> K = 16*(e>>3) + 8*half + (e&7)
// => two contiguous 8-half runs per lane at kOff+8*half and kOff+16+8*half.
__device__ __forceinline__ v16bf load_A_frag(const __bf16* base, int ldHalfs, int kOff) {
  int lane = threadIdx.x & 31;
  int half = lane >> 4;
  int m    = lane & 15;
  const __bf16* p = base + (size_t)m * ldHalfs + kOff + half * 8;
  v8bf lo = *(const v8bf*)p;
  v8bf hi = *(const v8bf*)(p + 16);
  return __builtin_shufflevector(lo, hi, 0,1,2,3,4,5,6,7,8,9,10,11,12,13,14,15);
}

// B 32x16 (KxN) sourced from row-major [n][K] memory (i.e. B = srcT):
// lane = {half,n}; element e <-> K = 16*(e>>3) + 8*half + (e&7) (same as A).
__device__ __forceinline__ v16bf load_B_fragT(const __bf16* base, int ldHalfs, int kOff) {
  int lane = threadIdx.x & 31;
  int half = lane >> 4;
  int n    = lane & 15;
  const __bf16* p = base + (size_t)n * ldHalfs + kOff + half * 8;
  v8bf lo = *(const v8bf*)p;
  v8bf hi = *(const v8bf*)(p + 16);
  return __builtin_shufflevector(lo, hi, 0,1,2,3,4,5,6,7,8,9,10,11,12,13,14,15);
}

__device__ __forceinline__ v8f wmma_bf16(v16bf a, v16bf b, v8f c) {
  return __builtin_amdgcn_wmma_f32_16x16x32_bf16(false, a, false, b, (short)0, c,
                                                 false, false);
}

// fast sigmoid: single v_rcp_f32 instead of precise-division sequence
__device__ __forceinline__ float fast_sigmoid(float x) {
  return __builtin_amdgcn_rcpf(1.f + __expf(-x));
}

// async copy 16B global -> LDS (per active lane), tracked by ASYNCcnt
__device__ __forceinline__ void async_copy_b128(unsigned lds_addr, const void* gptr) {
  asm volatile("global_load_async_to_lds_b128 %0, %1, off"
               :: "v"(lds_addr), "v"(gptr) : "memory");
}
__device__ __forceinline__ void wait_async0() {
  asm volatile("s_wait_asynccnt 0" ::: "memory");
}

// ---- Kernel A0: user profile = 0.25*(user+gender+age+occup) ---------------
__global__ __launch_bounds__(256)
void upmim_profile(const int* __restrict__ uid, const int* __restrict__ age,
                   const int* __restrict__ gender, const int* __restrict__ occup,
                   const float* __restrict__ ut, const float* __restrict__ at,
                   const float* __restrict__ gt, const float* __restrict__ ot,
                   float* __restrict__ up) {
  int i = blockIdx.x * blockDim.x + threadIdx.x;      // B*D
  if (i >= B_ * D_) return;
  int b = i >> 6, d = i & 63;
  up[i] = 0.25f * (ut[(size_t)uid[b]    * D_ + d] +
                   gt[(size_t)gender[b] * D_ + d] +
                   at[(size_t)age[b]    * D_ + d] +
                   ot[(size_t)occup[b]  * D_ + d]);
}

// ---- Kernel A1: stage item_his (masked) and hu = item*profile as bf16 ------
// Rows s in [S,SPAD) zero-filled so WMMA tiles run with full EXEC.
__global__ __launch_bounds__(256)
void upmim_stage(const int* __restrict__ mid_his, const float* __restrict__ mask,
                 const float* __restrict__ memb, const float* __restrict__ up,
                 __bf16* __restrict__ item, __bf16* __restrict__ hu) {
  int i = blockIdx.x * blockDim.x + threadIdx.x;      // B*SPAD*D
  if (i >= B_ * SPAD * D_) return;
  int d  = i & 63;
  int bs = i >> 6;
  int s  = bs % SPAD;
  int b  = bs / SPAD;
  float v = 0.f;
  if (s < S_) {
    int mid = mid_his[b * S_ + s];
    v = memb[(size_t)mid * D_ + d] * mask[b * S_ + s];
  }
  item[i] = (__bf16)v;
  hu[i]   = (__bf16)(v * up[b * D_ + d]);
}

// ---- Kernel A2: w_capsule f32 -> bf16 --------------------------------------
__global__ __launch_bounds__(256)
void upmim_castw(const float* __restrict__ wf, __bf16* __restrict__ wb) {
  int i = blockIdx.x * blockDim.x + threadIdx.x;      // S*E*D
  if (i < S_ * E_ * D_) wb[i] = (__bf16)wf[i];
}

// ---- Kernel B: adj = sigmoid(hu @ item^T) * mask_outer ---------------------
// Block (b, rowTile); 8 waves sweep 13 column tiles; K=64 -> 2 chained WMMAs.
// S = 200 = 12*16 + 8, so a lane's 8 rows are OOB iff (ti==12 && half==1):
// one uniform predicate, no per-row EXEC cascade.
__global__ __launch_bounds__(256)
void upmim_adj(const __bf16* __restrict__ hu, const __bf16* __restrict__ item,
               const float* __restrict__ mask, float* __restrict__ adj) {
  int b    = blockIdx.x;
  int ti   = blockIdx.y;                              // 0..12
  int wave = threadIdx.x >> 5;
  int lane = threadIdx.x & 31;
  int half = lane >> 4, n = lane & 15;

  const __bf16* Abase = hu + ((size_t)b * SPAD + ti * 16) * D_;
  v16bf a0 = load_A_frag(Abase, D_, 0);
  v16bf a1 = load_A_frag(Abase, D_, 32);

  const bool rowsOK = (ti < 12) | (half == 0);
  const int  M0     = ti * 16 + half * 8;             // lane's first row
  float rowm[8];
  #pragma unroll
  for (int r = 0; r < 8; ++r)
    rowm[r] = rowsOK ? mask[b * S_ + M0 + r] : 0.f;

  float* adjRow0 = adj + (size_t)b * S_ * S_ + (size_t)M0 * S_;

  for (int tj = wave; tj < 13; tj += 8) {
    const __bf16* Bbase = item + ((size_t)b * SPAD + tj * 16) * D_;
    v16bf b0 = load_B_fragT(Bbase, D_, 0);
    v16bf b1 = load_B_fragT(Bbase, D_, 32);
    v8f c = {};
    c = wmma_bf16(a0, b0, c);
    c = wmma_bf16(a1, b1, c);

    int N = tj * 16 + n;
    if (rowsOK && N < S_) {
      float mN = mask[b * S_ + N];
      #pragma unroll
      for (int r = 0; r < 8; ++r)
        adjRow0[(size_t)r * S_ + N] = fast_sigmoid(c[r]) * rowm[r] * mN;
    }
  }
}

// ---- Kernel C: hat[b,k,s,d] -- per-s GEMM (16 batch rows)x(64)x(256) -------
__global__ __launch_bounds__(256)
void upmim_hat(const __bf16* __restrict__ item, const __bf16* __restrict__ wcap,
               __bf16* __restrict__ hat) {
  int s    = blockIdx.x;                              // 0..S-1
  int b0   = blockIdx.y * 16;
  int wave = threadIdx.x >> 5;
  int lane = threadIdx.x & 31;
  int half = lane >> 4, n = lane & 15;

  // A rows m -> batch b0+m, K = d; row stride SPAD*D halfs
  const __bf16* Abase = item + ((size_t)b0 * SPAD + s) * D_;
  v16bf a0 = load_A_frag(Abase, SPAD * D_, 0);
  v16bf a1 = load_A_frag(Abase, SPAD * D_, 32);

  for (int tj = wave; tj < 16; tj += 8) {
    const __bf16* Bbase = wcap + ((size_t)s * E_ + tj * 16) * D_;
    v16bf b0v = load_B_fragT(Bbase, D_, 0);
    v16bf b1v = load_B_fragT(Bbase, D_, 32);
    v8f c = {};
    c = wmma_bf16(a0, b0v, c);
    c = wmma_bf16(a1, b1v, c);

    int e = tj * 16 + n;
    int k = e >> 6, dd = e & 63;
    #pragma unroll
    for (int r = 0; r < 8; ++r) {
      int bb = b0 + half * 8 + r;
      hat[(((size_t)bb * NI_ + k) * S_ + s) * D_ + dd] = (__bf16)c[r];
    }
  }
}

// ---- Kernel D: dynamic routing, one workgroup per b, hat[b] kept in LDS ----
__global__ __launch_bounds__(256)
void upmim_routing(const __bf16* __restrict__ hat, const float* __restrict__ mask,
                   float* __restrict__ cap_out) {
  extern __shared__ char smem[];
  __bf16* s_hat  = (__bf16*)smem;                          // NI*S*D halfs (100KB)
  float*  s_mask = (float*)(smem + (size_t)NI_ * S_ * D_ * 2);  // S
  float*  s_cw   = s_mask + S_;                            // NI*S
  float*  s_sw   = s_cw + NI_ * S_;                        // NI*S
  float*  s_cap  = s_sw + NI_ * S_;                        // NI*D
  float*  s_red  = s_cap + NI_ * D_;                       // 256
  float*  s_n2   = s_red + 256;                            // NI

  int b = blockIdx.x;
  int t = threadIdx.x;                                     // 0..255

  // async copy hat[b] (100KB) straight into LDS, 16B per lane per issue
  {
    const char* g = (const char*)(hat + (size_t)b * NI_ * S_ * D_);
    unsigned lds0 = (unsigned)(uintptr_t)(void*)s_hat;     // LDS byte offset
    const int n16 = NI_ * S_ * D_ * 2 / 16;                // 6400 chunks
    for (int i = t; i < n16; i += 256)
      async_copy_b128(lds0 + i * 16, g + (size_t)i * 16);
  }
  for (int i = t; i < S_; i += 256)       s_mask[i] = mask[b * S_ + i];
  for (int i = t; i < NI_ * S_; i += 256) s_cw[i]   = 0.f;
  wait_async0();
  __syncthreads();

  int k = t >> 6, d = t & 63;

  for (int it = 0; it < 3; ++it) {
    // softmax over NI per position, then mask
    for (int s = t; s < S_; s += 256) {
      float c0 = s_cw[0 * S_ + s], c1 = s_cw[1 * S_ + s];
      float c2 = s_cw[2 * S_ + s], c3 = s_cw[3 * S_ + s];
      float mx = fmaxf(fmaxf(c0, c1), fmaxf(c2, c3));
      float e0 = __expf(c0 - mx), e1 = __expf(c1 - mx);
      float e2 = __expf(c2 - mx), e3 = __expf(c3 - mx);
      float inv = __builtin_amdgcn_rcpf(e0 + e1 + e2 + e3);
      float mk  = (s_mask[s] == 0.f) ? 0.f : 1.f;
      s_sw[0 * S_ + s] = e0 * inv * mk;
      s_sw[1 * S_ + s] = e1 * inv * mk;
      s_sw[2 * S_ + s] = e2 * inv * mk;
      s_sw[3 * S_ + s] = e3 * inv * mk;
    }
    __syncthreads();

    // cap[k,d] = sum_s sw[k,s] * hat[k,s,d]
    float acc = 0.f;
    for (int s = 0; s < S_; ++s)
      acc += s_sw[k * S_ + s] * (float)s_hat[((size_t)k * S_ + s) * D_ + d];

    // squash: n/(1+n)/sqrt(n+eps) * v
    s_red[t] = acc * acc;
    __syncthreads();
    if (t < NI_) {
      float n2 = 0.f;
      for (int j = 0; j < D_; ++j) n2 += s_red[t * D_ + j];
      s_n2[t] = n2;
    }
    __syncthreads();
    float n2    = s_n2[k];
    float scale = n2 * __builtin_amdgcn_rcpf(1.f + n2) *
                  __builtin_amdgcn_rsqf(n2 + 1e-9f);
    s_cap[t] = acc * scale;
    __syncthreads();

    if (it < 2) {
      // cw[k,s] += dot_d(hat[k,s,:], cap[k,:])
      for (int p = t; p < NI_ * S_; p += 256) {
        int kk = p / S_, ss = p % S_;
        float dot = 0.f;
        for (int dd = 0; dd < D_; ++dd)
          dot += (float)s_hat[((size_t)kk * S_ + ss) * D_ + dd] * s_cap[kk * D_ + dd];
        s_cw[p] += dot;
      }
      __syncthreads();
    }
  }
  cap_out[(size_t)b * NI_ * D_ + t] = s_cap[t];
}

// ---------------------------------------------------------------------------
extern "C" void kernel_launch(void* const* d_in, const int* in_sizes, int n_in,
                              void* d_out, int out_size, void* d_ws, size_t ws_size,
                              hipStream_t stream) {
  (void)in_sizes; (void)n_in; (void)out_size; (void)ws_size;

  const int*   uid    = (const int*)d_in[0];
  const int*   age    = (const int*)d_in[1];
  const int*   gender = (const int*)d_in[2];
  const int*   occup  = (const int*)d_in[3];
  const int*   midhis = (const int*)d_in[4];
  const float* mask   = (const float*)d_in[5];
  const float* ut     = (const float*)d_in[6];
  const float* at     = (const float*)d_in[7];
  const float* gt     = (const float*)d_in[8];
  const float* ot     = (const float*)d_in[9];
  const float* memb   = (const float*)d_in[10];
  const float* wcapf  = (const float*)d_in[11];

  // workspace layout (all offsets 32B-aligned):
  char*   w     = (char*)d_ws;
  float*  up    = (float*)w;                                   // B*D f32
  __bf16* item  = (__bf16*)(w + (size_t)B_ * D_ * 4);          // B*SPAD*D bf16
  __bf16* hu    = item + (size_t)B_ * SPAD * D_;               // B*SPAD*D bf16
  __bf16* wcapb = hu   + (size_t)B_ * SPAD * D_;               // S*E*D bf16
  __bf16* hatw  = wcapb + (size_t)S_ * E_ * D_;                // B*NI*S*D bf16

  float* cap_out = (float*)d_out;                              // B*NI*D
  float* adj_out = cap_out + (size_t)B_ * NI_ * D_;            // B*S*S

  upmim_profile<<<(B_ * D_) / 256, 256, 0, stream>>>(uid, age, gender, occup,
                                                     ut, at, gt, ot, up);
  upmim_stage<<<(B_ * SPAD * D_) / 256, 256, 0, stream>>>(midhis, mask, memb,
                                                          up, item, hu);
  upmim_castw<<<(S_ * E_ * D_) / 256, 256, 0, stream>>>(wcapf, wcapb);

  upmim_adj<<<dim3(B_, 13), 256, 0, stream>>>(hu, item, mask, adj_out);
  upmim_hat<<<dim3(S_, B_ / 16), 256, 0, stream>>>(item, wcapb, hatw);

  size_t smem = (size_t)NI_ * S_ * D_ * 2 +
                (S_ + 2 * NI_ * S_ + NI_ * D_ + 256 + NI_) * sizeof(float);
  upmim_routing<<<B_, 256, smem, stream>>>(hatw, mask, cap_out);
}
